// peephole_LSTM_10307921510949
// MI455X (gfx1250) — matrix operand order repaired and checked by reading, not compiled, not run on hardware
//
#include <hip/hip_runtime.h>

// ---------------------------------------------------------------------------
// Peephole LSTM on gfx1250 (MI455X): f16 WMMA GEMMs + fp32 state.
// ---------------------------------------------------------------------------

#define VOCAB  6000
#define HIDDEN 1024
#define SEQ    128
#define BATCH  64

#define NT_H   (HIDDEN / 16)       // 64 N-tiles over hidden
#define NT_V   (VOCAB / 16)        // 375 N-tiles over vocab
#define KC_REC (2 * HIDDEN / 32)   // 64 K-chunks over [h ; c] (K = 2048)
#define KC_OUT (HIDDEN / 32)       // 32 K-chunks over hidden
#define BH     (BATCH * HIDDEN)    // 65536 elements per time slice

typedef __attribute__((ext_vector_type(16))) _Float16 v16h;
typedef __attribute__((ext_vector_type(8)))  _Float16 v8h;
typedef __attribute__((ext_vector_type(8)))  float    v8f;

union AFrag { v16h v; v8h h[2]; };

__device__ __forceinline__ float sigmoidf_fast(float x) {
    return 1.0f / (1.0f + __expf(-x));
}

// ---------------------------------------------------------------------------
// Pack recurrent weights [Wh_g ; Wc_g] (each 1024x1024 fp32, row-major K x N)
// into f16 B-fragment layout:
//   dst[((gate*64 + nt)*64 + kc)*512 + lane*16 + h]
//   lane holds column n = nt*16 + (lane & 15); half h <-> K = (h&7) + 8*(lane>>4) + 16*(h>>3)
// ---------------------------------------------------------------------------
__global__ __launch_bounds__(256) void pack_rec_kernel(
    const float* __restrict__ Whf, const float* __restrict__ Whi,
    const float* __restrict__ Who, const float* __restrict__ Whc,
    const float* __restrict__ Wcf, const float* __restrict__ Wci,
    const float* __restrict__ Wco, const float* __restrict__ Wcc,
    _Float16* __restrict__ out)
{
    unsigned idx = blockIdx.x * 256u + threadIdx.x;   // 4*64*64*512 = 8388608
    unsigned hh   = idx & 15u;
    unsigned lane = (idx >> 4) & 31u;
    unsigned kc   = (idx >> 9) & 63u;
    unsigned nt   = (idx >> 15) & 63u;
    unsigned gate = idx >> 21;

    unsigned Kl = (hh & 7u) + 8u * (lane >> 4) + 16u * (hh >> 3);
    unsigned n  = nt * 16u + (lane & 15u);
    unsigned k  = kc * 32u + Kl;              // 0..2047

    const float* Wh[4] = {Whf, Whi, Who, Whc};
    const float* Wc[4] = {Wcf, Wci, Wco, Wcc};
    const float* src = (k < HIDDEN) ? Wh[gate] : Wc[gate];
    unsigned krow = (k < HIDDEN) ? k : (k - HIDDEN);
    out[idx] = (_Float16)src[(size_t)krow * HIDDEN + n];
}

// Pack W_out (1024 x 6000 fp32) the same way:
//   dst[(nt*32 + kc)*512 + lane*16 + h]
__global__ __launch_bounds__(256) void pack_out_kernel(
    const float* __restrict__ Wout, _Float16* __restrict__ out)
{
    unsigned idx = blockIdx.x * 256u + threadIdx.x;   // 375*32*512 = 6144000
    if (idx >= (unsigned)(NT_V * KC_OUT * 512)) return;
    unsigned hh   = idx & 15u;
    unsigned lane = (idx >> 4) & 31u;
    unsigned kc   = (idx >> 9) & 31u;
    unsigned nt   = idx >> 14;

    unsigned Kl = (hh & 7u) + 8u * (lane >> 4) + 16u * (hh >> 3);
    unsigned n  = nt * 16u + (lane & 15u);
    unsigned k  = kc * 32u + Kl;
    out[idx] = (_Float16)Wout[(size_t)k * VOCAB + n];
}

// Initialize carried state: hseq[0] = f16(h0), c16 = f16(c0), c32 = c0.
__global__ __launch_bounds__(256) void init_state_kernel(
    const float* __restrict__ h0, const float* __restrict__ c0,
    _Float16* __restrict__ hseq, _Float16* __restrict__ c16,
    float* __restrict__ c32)
{
    unsigned i = blockIdx.x * 256u + threadIdx.x;
    if (i >= BH) return;
    hseq[i] = (_Float16)h0[i];
    c16[i]  = (_Float16)c0[i];
    c32[i]  = c0[i];
}

// ---------------------------------------------------------------------------
// One LSTM timestep.
// grid = (64 n-tiles, 4 m-tiles), block = 128 (4 waves; wave == gate).
// Each wave computes a 16x16 pre-activation tile with 64 x v_wmma over K=2048
// ([h ; c]), waves exchange gates through LDS, then the block does the fused
// gather (input projection) + gate nonlinearities + state update.
// ---------------------------------------------------------------------------
__global__ __launch_bounds__(128) void lstm_step_kernel(
    int t,
    const int*      __restrict__ tokens,
    _Float16*       __restrict__ hseq,     // (SEQ+1) x BATCH x HIDDEN f16
    const _Float16* __restrict__ c16_r,
    _Float16*       __restrict__ c16_w,
    const float*    __restrict__ c32_r,
    float*          __restrict__ c32_w,
    float*          __restrict__ h32,
    const _Float16* __restrict__ Wpack,    // [4][64][64][512] f16
    const float* __restrict__ W_if, const float* __restrict__ W_ii,
    const float* __restrict__ W_io, const float* __restrict__ W_ic,
    const float* __restrict__ b_f,  const float* __restrict__ b_i,
    const float* __restrict__ b_o,  const float* __restrict__ b_c)
{
    const int nt    = blockIdx.x;          // hidden n-tile (0..63)
    const int mtile = blockIdx.y;          // batch m-tile (0..3)
    const int gate  = threadIdx.x >> 5;    // wave id = gate id
    const int lane  = threadIdx.x & 31;
    const int g     = lane >> 4;           // lane group
    const int row   = mtile * 16 + (lane & 15);

    __shared__ float pre[4][256];

    const _Float16* __restrict__ hA = hseq + (size_t)t * BH;
    const _Float16* __restrict__ Bp =
        Wpack + (((size_t)gate * NT_H + nt) * KC_REC) * 512 + (size_t)lane * 16;

    v8f acc = {};

    // --- K = 0..1023 : h @ Wh_gate ---
    #pragma unroll 4
    for (int kc = 0; kc < 32; ++kc) {
        const int kb = kc * 32;
        AFrag a;
        a.h[0] = *(const v8h*)(hA + (size_t)row * HIDDEN + kb + 8 * g);
        a.h[1] = *(const v8h*)(hA + (size_t)row * HIDDEN + kb + 16 + 8 * g);
        v16h b = *(const v16h*)(Bp + (size_t)kc * 512);
        acc = __builtin_amdgcn_wmma_f32_16x16x32_f16(
            false, a.v, false, b, (short)0, acc, false, false);
    }
    // --- K = 1024..2047 : c @ Wc_gate ---
    #pragma unroll 4
    for (int kc = 0; kc < 32; ++kc) {
        const int kb = kc * 32;
        AFrag a;
        a.h[0] = *(const v8h*)(c16_r + (size_t)row * HIDDEN + kb + 8 * g);
        a.h[1] = *(const v8h*)(c16_r + (size_t)row * HIDDEN + kb + 16 + 8 * g);
        v16h b = *(const v16h*)(Bp + (size_t)(32 + kc) * 512);
        acc = __builtin_amdgcn_wmma_f32_16x16x32_f16(
            false, a.v, false, b, (short)0, acc, false, false);
    }

    // Spill 16x16 tile: C/D layout -> lane l, vgpr j : M = j + 8*(l>>4), N = l&15
    #pragma unroll
    for (int j = 0; j < 8; ++j) {
        int m_l = j + 8 * g;
        pre[gate][m_l * 16 + (lane & 15)] = acc[j];
    }
    __syncthreads();

    // Fused gather + gate math + state update (256 elems / 128 threads).
    #pragma unroll
    for (int u = 0; u < 2; ++u) {
        const int e    = threadIdx.x * 2 + u;
        const int m_l  = e >> 4;
        const int cl   = e & 15;
        const int grow = mtile * 16 + m_l;        // batch row
        const int col  = nt * 16 + cl;            // hidden col
        const size_t sidx = (size_t)grow * HIDDEN + col;

        const int tok = tokens[(size_t)t * BATCH + grow];
        const size_t toff = (size_t)tok * HIDDEN + col;

        float pf = pre[0][e] + W_if[toff] + b_f[col];
        float pi = pre[1][e] + W_ii[toff] + b_i[col];
        float po = pre[2][e] + W_io[toff] + b_o[col];
        float pc = pre[3][e] + W_ic[toff] + b_c[col];

        float f    = sigmoidf_fast(pf);
        float ig   = sigmoidf_fast(pi);
        float o    = sigmoidf_fast(po);
        float cand = tanhf(pc);

        float c_new = f * c32_r[sidx] + ig * cand;
        float h_new = o * tanhf(c_new);

        c32_w[sidx] = c_new;
        c16_w[sidx] = (_Float16)c_new;
        h32[sidx]   = h_new;
        hseq[(size_t)(t + 1) * BH + sidx] = (_Float16)h_new;
    }
}

// ---------------------------------------------------------------------------
// logits[s,b,v] = hseq[s+1] @ W_out + b_out.
// grid = (375 n-tiles, 32 row-blocks), block = 256 (8 waves).
// Each wave: 2 M-tiles (32 rows) x 16 cols, K = 1024 -> 64 v_wmma.
// ---------------------------------------------------------------------------
__global__ __launch_bounds__(256) void logits_kernel(
    const _Float16* __restrict__ hseq,     // base (includes t=0 slot)
    const _Float16* __restrict__ WoutP,    // packed [375][32][512]
    const float*    __restrict__ b_out,
    float*          __restrict__ out)      // SEQ*BATCH x VOCAB
{
    const int nt   = blockIdx.x;           // vocab n-tile (0..374)
    const int mb   = blockIdx.y;           // row-block (0..31), 256 rows each
    const int wave = threadIdx.x >> 5;
    const int lane = threadIdx.x & 31;
    const int g    = lane >> 4;

    const int rowbase = mb * 256 + wave * 32;
    const int r0 = rowbase + (lane & 15);
    const int r1 = r0 + 16;

    const _Float16* __restrict__ A  = hseq + BH;   // skip initial state slot
    const _Float16* __restrict__ Bp =
        WoutP + (size_t)nt * KC_OUT * 512 + (size_t)lane * 16;

    v8f acc0 = {}, acc1 = {};
    #pragma unroll 4
    for (int kc = 0; kc < KC_OUT; ++kc) {
        const int kb = kc * 32;
        AFrag a0, a1;
        a0.h[0] = *(const v8h*)(A + (size_t)r0 * HIDDEN + kb + 8 * g);
        a0.h[1] = *(const v8h*)(A + (size_t)r0 * HIDDEN + kb + 16 + 8 * g);
        a1.h[0] = *(const v8h*)(A + (size_t)r1 * HIDDEN + kb + 8 * g);
        a1.h[1] = *(const v8h*)(A + (size_t)r1 * HIDDEN + kb + 16 + 8 * g);
        v16h b = *(const v16h*)(Bp + (size_t)kc * 512);
        acc0 = __builtin_amdgcn_wmma_f32_16x16x32_f16(
            false, a0.v, false, b, (short)0, acc0, false, false);
        acc1 = __builtin_amdgcn_wmma_f32_16x16x32_f16(
            false, a1.v, false, b, (short)0, acc1, false, false);
    }

    const int col = nt * 16 + (lane & 15);
    const float bo = b_out[col];
    #pragma unroll
    for (int j = 0; j < 8; ++j) {
        int m0 = rowbase + j + 8 * g;
        out[(size_t)m0 * VOCAB + col]        = acc0[j] + bo;
        out[(size_t)(m0 + 16) * VOCAB + col] = acc1[j] + bo;
    }
}

// Emit final (h_T, c_T) as fp32 after the logits block.
__global__ __launch_bounds__(256) void tail_kernel(
    const float* __restrict__ h32, const float* __restrict__ c32,
    float* __restrict__ outH, float* __restrict__ outC)
{
    unsigned i = blockIdx.x * 256u + threadIdx.x;
    if (i >= BH) return;
    outH[i] = h32[i];
    outC[i] = c32[i];
}

// ---------------------------------------------------------------------------
extern "C" void kernel_launch(void* const* d_in, const int* in_sizes, int n_in,
                              void* d_out, int out_size, void* d_ws, size_t ws_size,
                              hipStream_t stream)
{
    (void)in_sizes; (void)n_in; (void)out_size; (void)ws_size;

    const int*   tokens = (const int*)  d_in[0];
    const float* h0     = (const float*)d_in[1];
    const float* c0     = (const float*)d_in[2];
    const float* W_if   = (const float*)d_in[3];
    const float* W_ii   = (const float*)d_in[4];
    const float* W_io   = (const float*)d_in[5];
    const float* W_ic   = (const float*)d_in[6];
    const float* b_f    = (const float*)d_in[7];
    const float* b_i    = (const float*)d_in[8];
    const float* b_o    = (const float*)d_in[9];
    const float* b_c    = (const float*)d_in[10];
    const float* Wh_f   = (const float*)d_in[11];
    const float* Wh_i   = (const float*)d_in[12];
    const float* Wh_o   = (const float*)d_in[13];
    const float* Wh_c   = (const float*)d_in[14];
    const float* Wc_f   = (const float*)d_in[15];
    const float* Wc_i   = (const float*)d_in[16];
    const float* Wc_o   = (const float*)d_in[17];
    const float* Wc_c   = (const float*)d_in[18];
    const float* W_out  = (const float*)d_in[19];
    const float* b_out  = (const float*)d_in[20];

    // Workspace layout
    char* ws = (char*)d_ws;
    size_t off = 0;
    _Float16* Wpack = (_Float16*)(ws + off); off += (size_t)4 * NT_H * KC_REC * 512 * 2; // 16 MB
    _Float16* WoutP = (_Float16*)(ws + off); off += (size_t)NT_V * KC_OUT * 512 * 2;     // ~12 MB
    _Float16* hseq  = (_Float16*)(ws + off); off += (size_t)(SEQ + 1) * BH * 2;          // ~17 MB
    _Float16* c16A  = (_Float16*)(ws + off); off += (size_t)BH * 2;
    _Float16* c16B  = (_Float16*)(ws + off); off += (size_t)BH * 2;
    float*    c32A  = (float*)   (ws + off); off += (size_t)BH * 4;
    float*    c32B  = (float*)   (ws + off); off += (size_t)BH * 4;
    float*    h32   = (float*)   (ws + off); off += (size_t)BH * 4;

    float* out_logits = (float*)d_out;
    float* out_hT     = out_logits + (size_t)SEQ * BATCH * VOCAB;
    float* out_cT     = out_hT + BH;

    // 1) Pack weights (f32 -> f16 fragment layout)
    pack_rec_kernel<<<(4u * NT_H * KC_REC * 512u) / 256u, 256, 0, stream>>>(
        Wh_f, Wh_i, Wh_o, Wh_c, Wc_f, Wc_i, Wc_o, Wc_c, Wpack);
    pack_out_kernel<<<((unsigned)(NT_V * KC_OUT * 512) + 255u) / 256u, 256, 0, stream>>>(
        W_out, WoutP);

    // 2) Initial state
    init_state_kernel<<<(BH + 255) / 256, 256, 0, stream>>>(h0, c0, hseq, c16A, c32A);

    // 3) Sequential recurrence
    for (int t = 0; t < SEQ; ++t) {
        const bool even = (t & 1) == 0;
        lstm_step_kernel<<<dim3(NT_H, BATCH / 16), 128, 0, stream>>>(
            t, tokens, hseq,
            even ? c16A : c16B, even ? c16B : c16A,
            even ? c32A : c32B, even ? c32B : c32A,
            h32, Wpack,
            W_if, W_ii, W_io, W_ic, b_f, b_i, b_o, b_c);
    }

    // 4) Output projection (all timesteps at once)
    logits_kernel<<<dim3(NT_V, (SEQ * BATCH) / 256), 256, 0, stream>>>(
        hseq, WoutP, b_out, out_logits);

    // 5) Final (h_T, c_T); after an even number of steps c lives in buffer A.
    tail_kernel<<<(BH + 255) / 256, 256, 0, stream>>>(h32, c32A, out_hT, out_cT);
}
